// VisionAttentionSink_49907519979570
// MI455X (gfx1250) — 16-bit and 8-bit matrix operand order repaired and checked by reading, compiled, not run on hardware
//
#include <hip/hip_runtime.h>
#include <hip/hip_bf16.h>
#include <cstdint>

typedef __attribute__((ext_vector_type(16))) _Float16 v16h;
typedef __attribute__((ext_vector_type(8)))  float    v8f;
typedef __attribute__((ext_vector_type(8)))  int      v8i;

#define NB   16
#define SEQ  1024
#define EMB  768
#define NH   12
#define HD   64
#define NTOK (NB * SEQ)
#define WELEM (EMB * EMB)
#define XELEM ((size_t)NTOK * EMB)

union FragH  { v16h v; int4 q[2]; };
union FragA8 { v8i  v; int2 d[4]; };
union FragB8 { v8i  v; int4 q[2]; };

__global__ void k_init_scalars(float* s) { s[threadIdx.x] = 0.0f; }

__global__ __launch_bounds__(256) void k_absmax_f32(const float* __restrict__ x,
                                                    size_t n, float* out) {
  __shared__ float sb[256];
  float m = 0.0f;
  for (size_t i = (size_t)blockIdx.x * blockDim.x + threadIdx.x; i < n;
       i += (size_t)gridDim.x * blockDim.x)
    m = fmaxf(m, fabsf(x[i]));
  sb[threadIdx.x] = m; __syncthreads();
  for (int off = 128; off > 0; off >>= 1) {
    if ((int)threadIdx.x < off) sb[threadIdx.x] = fmaxf(sb[threadIdx.x], sb[threadIdx.x + off]);
    __syncthreads();
  }
  if (threadIdx.x == 0) atomicMax((unsigned int*)out, __float_as_uint(sb[0]));
}

__global__ __launch_bounds__(256) void k_absmax_f16(const _Float16* __restrict__ x,
                                                    size_t n, float* out) {
  __shared__ float sb[256];
  float m = 0.0f;
  for (size_t i = (size_t)blockIdx.x * blockDim.x + threadIdx.x; i < n;
       i += (size_t)gridDim.x * blockDim.x)
    m = fmaxf(m, fabsf((float)x[i]));
  sb[threadIdx.x] = m; __syncthreads();
  for (int off = 128; off > 0; off >>= 1) {
    if ((int)threadIdx.x < off) sb[threadIdx.x] = fmaxf(sb[threadIdx.x], sb[threadIdx.x + off]);
    __syncthreads();
  }
  if (threadIdx.x == 0) atomicMax((unsigned int*)out, __float_as_uint(sb[0]));
}

__global__ __launch_bounds__(256) void k_abssum_part(const float* w0, const float* w1,
                                                     const float* w2, const float* w3,
                                                     int n, float* partials) {
  __shared__ float sb[256];
  const float* w = (blockIdx.y == 0) ? w0 : (blockIdx.y == 1) ? w1
                   : (blockIdx.y == 2) ? w2 : w3;
  float acc = 0.0f;
  for (int i = blockIdx.x * blockDim.x + threadIdx.x; i < n; i += 64 * 256)
    acc += fabsf(w[i]);
  sb[threadIdx.x] = acc; __syncthreads();
  for (int off = 128; off > 0; off >>= 1) {
    if ((int)threadIdx.x < off) sb[threadIdx.x] += sb[threadIdx.x + off];
    __syncthreads();
  }
  if (threadIdx.x == 0) partials[blockIdx.y * 64 + blockIdx.x] = sb[0];
}

__global__ void k_sum_final(const float* partials, float* s) {
  __shared__ float sb[64];
  sb[threadIdx.x] = partials[blockIdx.x * 64 + threadIdx.x];
  __syncthreads();
  for (int off = 32; off > 0; off >>= 1) {
    if ((int)threadIdx.x < off) sb[threadIdx.x] += sb[threadIdx.x + off];
    __syncthreads();
  }
  if (threadIdx.x == 0) s[1 + blockIdx.x] = sb[0];
}

__global__ __launch_bounds__(256) void k_quant_x(const float* __restrict__ x,
                                                 const float* __restrict__ scal,
                                                 int8_t* __restrict__ qx, size_t n) {
  const float isc = fmaxf(scal[0], 1e-8f) * (1.0f / 127.0f);
  const float inv = 1.0f / isc;
  for (size_t i = (size_t)blockIdx.x * blockDim.x + threadIdx.x; i < n;
       i += (size_t)gridDim.x * blockDim.x) {
    float q = rintf(x[i] * inv);
    q = fminf(fmaxf(q, -128.0f), 127.0f);
    qx[i] = (int8_t)q;
  }
}

__global__ __launch_bounds__(256) void k_quant_w(const float* w0, const float* w1,
                                                 const float* w2, const float* w3,
                                                 const float* __restrict__ scal,
                                                 int8_t* __restrict__ qw) {
  const int wi = blockIdx.y;
  const float* w = (wi == 0) ? w0 : (wi == 1) ? w1 : (wi == 2) ? w2 : w3;
  const float wsc = fmaxf(scal[1 + wi] * (1.0f / (float)WELEM), 1e-5f);
  const float thr = 0.5f * wsc;
  int i = blockIdx.x * 256 + threadIdx.x;
  float v = w[i];
  qw[(size_t)wi * WELEM + i] = (int8_t)((v > thr) ? 1 : (v < -thr ? -1 : 0));
}

__global__ __launch_bounds__(256) void k_quant_o(const _Float16* __restrict__ o1,
                                                 const float* __restrict__ scal,
                                                 int8_t* __restrict__ qa, size_t n) {
  const float isc = fmaxf(scal[5], 1e-8f) * (1.0f / 127.0f);
  const float inv = 1.0f / isc;
  for (size_t i = (size_t)blockIdx.x * blockDim.x + threadIdx.x; i < n;
       i += (size_t)gridDim.x * blockDim.x) {
    float q = rintf((float)o1[i] * inv);
    q = fminf(fmaxf(q, -128.0f), 127.0f);
    qa[i] = (int8_t)q;
  }
}

__device__ __forceinline__ void gemm_i8_tile(const int8_t* __restrict__ qx,
                                             const int8_t* __restrict__ qw,
                                             int mbase, int nbase, int lane,
                                             v8i acc[2][2]) {
  const int rlo    = lane & 15;
  const int khalfA = (lane >> 4) << 4;
  const int khalfB = (lane >> 4) << 4;
  const int8_t* arow0 = qx + (size_t)(mbase + rlo) * EMB;
  const int8_t* arow1 = arow0 + (size_t)16 * EMB;
  const int8_t* brow0 = qw + (size_t)(nbase + rlo) * EMB;
  const int8_t* brow1 = brow0 + (size_t)16 * EMB;
  for (int kk = 0; kk < EMB; kk += 64) {
    if (kk + 64 < EMB) {
      __builtin_prefetch(arow0 + kk + 64, 0, 0);
      __builtin_prefetch(brow0 + kk + 64, 0, 0);
    }
    FragA8 a0, a1; FragB8 b0, b1;
    const int ab = kk + khalfA;
#pragma unroll
    for (int t = 0; t < 4; ++t) {
      a0.d[t] = *(const int2*)(arow0 + ab + 8 * t + 16 * (t >> 1));
      a1.d[t] = *(const int2*)(arow1 + ab + 8 * t + 16 * (t >> 1));
    }
    const int bb = kk + khalfB;
    b0.q[0] = *(const int4*)(brow0 + bb);
    b0.q[1] = *(const int4*)(brow0 + bb + 32);
    b1.q[0] = *(const int4*)(brow1 + bb);
    b1.q[1] = *(const int4*)(brow1 + bb + 32);
    acc[0][0] = __builtin_amdgcn_wmma_i32_16x16x64_iu8(true, a0.v, true, b0.v, acc[0][0], false, false);
    acc[0][1] = __builtin_amdgcn_wmma_i32_16x16x64_iu8(true, a0.v, true, b1.v, acc[0][1], false, false);
    acc[1][0] = __builtin_amdgcn_wmma_i32_16x16x64_iu8(true, a1.v, true, b0.v, acc[1][0], false, false);
    acc[1][1] = __builtin_amdgcn_wmma_i32_16x16x64_iu8(true, a1.v, true, b1.v, acc[1][1], false, false);
  }
}

__global__ __launch_bounds__(256) void k_gemm_qkv(
    const int8_t* __restrict__ qx, const int8_t* __restrict__ qw_all,
    const float* __restrict__ bq, const float* __restrict__ bk,
    const float* __restrict__ bv, const float* __restrict__ scal,
    _Float16* __restrict__ Qh, _Float16* __restrict__ Kh, _Float16* __restrict__ Vt) {
  const int lane  = threadIdx.x & 31;
  const int wave  = threadIdx.x >> 5;
  const int z     = blockIdx.z;
  const int mbase = blockIdx.x * 128 + (wave >> 1) * 32;
  const int nbase = blockIdx.y * 64 + (wave & 1) * 32;
  v8i acc[2][2] = {};
  gemm_i8_tile(qx, qw_all + (size_t)z * WELEM, mbase, nbase, lane, acc);

  const float wsc = fmaxf(scal[1 + z] * (1.0f / (float)WELEM), 1e-5f);
  const float isc = fmaxf(scal[0], 1e-8f) * (1.0f / 127.0f);
  const float sc  = wsc * isc;
  const float* bias = (z == 0) ? bq : (z == 1 ? bk : bv);
  const int mloc = (lane >> 4) << 3;
  const int nloc = lane & 15;
#pragma unroll
  for (int im = 0; im < 2; ++im)
#pragma unroll
    for (int jn = 0; jn < 2; ++jn)
#pragma unroll
      for (int j = 0; j < 8; ++j) {
        int m = mbase + 16 * im + mloc + j;
        int o = nbase + 16 * jn + nloc;
        float val = (float)acc[im][jn][j] * sc + bias[o];
        int bt = m >> 10, sp = m & 1023;
        int h = o >> 6, d = o & 63;
        if (z == 0)
          Qh[(((size_t)bt * NH + h) * SEQ + sp) * HD + d] = (_Float16)val;
        else if (z == 1)
          Kh[(((size_t)bt * NH + h) * SEQ + sp) * HD + d] = (_Float16)val;
        else
          Vt[(((size_t)bt * NH + h) * HD + d) * SEQ + sp] = (_Float16)val;
      }
}

__global__ __launch_bounds__(256) void k_gemm_o(
    const int8_t* __restrict__ qa, const int8_t* __restrict__ qw,
    const float* __restrict__ bo, const float* __restrict__ scal,
    float* __restrict__ out) {
  const int lane  = threadIdx.x & 31;
  const int wave  = threadIdx.x >> 5;
  const int mbase = blockIdx.x * 128 + (wave >> 1) * 32;
  const int nbase = blockIdx.y * 64 + (wave & 1) * 32;
  v8i acc[2][2] = {};
  gemm_i8_tile(qa, qw, mbase, nbase, lane, acc);

  const float wsc = fmaxf(scal[4] * (1.0f / (float)WELEM), 1e-5f);
  const float isc = fmaxf(scal[5], 1e-8f) * (1.0f / 127.0f);
  const float sc  = wsc * isc;
  const int mloc = (lane >> 4) << 3;
  const int nloc = lane & 15;
#pragma unroll
  for (int im = 0; im < 2; ++im)
#pragma unroll
    for (int jn = 0; jn < 2; ++jn)
#pragma unroll
      for (int j = 0; j < 8; ++j) {
        int m = mbase + 16 * im + mloc + j;
        int o = nbase + 16 * jn + nloc;
        out[(size_t)m * EMB + o] = (float)acc[im][jn][j] * sc + bo[o];
      }
}

__global__ __launch_bounds__(256) void k_flash(
    const _Float16* __restrict__ Qh, const _Float16* __restrict__ Kh,
    const _Float16* __restrict__ Vt, _Float16* __restrict__ O1) {
  __shared__ __align__(16) _Float16 Plds[8][512];
  const int lane  = threadIdx.x & 31;
  const int wave  = threadIdx.x >> 5;
  const int bh    = blockIdx.y;
  const int b     = bh / NH, h = bh % NH;
  const int qbase = blockIdx.x * 128 + wave * 16;
  const _Float16* Qb = Qh + (size_t)bh * SEQ * HD;
  const _Float16* Kb = Kh + (size_t)bh * SEQ * HD;
  const _Float16* Vb = Vt + (size_t)bh * HD * SEQ;

  const int rlo = lane & 15;
  const int hb8 = (lane >> 4) << 3;
  const int hb16 = (lane >> 4) << 4;

  FragH qf0, qf1;
  {
    const _Float16* qp = Qb + (size_t)(qbase + rlo) * HD;
    qf0.q[0] = *(const int4*)(qp + hb8);
    qf0.q[1] = *(const int4*)(qp + hb8 + 16);
    qf1.q[0] = *(const int4*)(qp + hb8 + 32);
    qf1.q[1] = *(const int4*)(qp + hb8 + 48);
  }

  float mrow[8], lrow[8];
#pragma unroll
  for (int j = 0; j < 8; ++j) { mrow[j] = -1e30f; lrow[j] = 0.0f; }
  v8f acc[4] = {};
  const float scale = 0.125f;
  _Float16* pw = &Plds[wave][0];

  for (int kt = 0; kt < SEQ; kt += 32) {
    v8f S0 = {}, S1 = {};
#pragma unroll
    for (int h2 = 0; h2 < 2; ++h2) {
      const _Float16* kp = Kb + (size_t)(kt + 16 * h2 + rlo) * HD + hb8;
      FragH klo, khi;
      klo.q[0] = *(const int4*)(kp);
      klo.q[1] = *(const int4*)(kp + 16);
      khi.q[0] = *(const int4*)(kp + 32);
      khi.q[1] = *(const int4*)(kp + 48);
      v8f Sc = {};
      Sc = __builtin_amdgcn_wmma_f32_16x16x32_f16(false, qf0.v, false, klo.v, (short)0, Sc, false, false);
      Sc = __builtin_amdgcn_wmma_f32_16x16x32_f16(false, qf1.v, false, khi.v, (short)0, Sc, false, false);
      if (h2 == 0) S0 = Sc; else S1 = Sc;
    }
#pragma unroll
    for (int j = 0; j < 8; ++j) {
      float x0 = S0[j] * scale;
      float x1 = S1[j] * scale;
      float mt = fmaxf(x0, x1);
#pragma unroll
      for (int msk = 8; msk >= 1; msk >>= 1)
        mt = fmaxf(mt, __shfl_xor(mt, msk, 16));
      float mnew = fmaxf(mrow[j], mt);
      float p0 = __expf(x0 - mnew);
      float p1 = __expf(x1 - mnew);
      float rs = p0 + p1;
#pragma unroll
      for (int msk = 8; msk >= 1; msk >>= 1)
        rs += __shfl_xor(rs, msk, 16);
      float alpha = __expf(mrow[j] - mnew);
      lrow[j] = lrow[j] * alpha + rs;
      mrow[j] = mnew;
#pragma unroll
      for (int t = 0; t < 4; ++t) acc[t][j] *= alpha;
      int qrow = j + hb8;
      pw[qrow * 32 + rlo]      = (_Float16)p0;
      pw[qrow * 32 + 16 + rlo] = (_Float16)p1;
    }
    FragH pf;
    {
      const _Float16* pr = pw + rlo * 32 + hb8;
      pf.q[0] = *(const int4*)(pr);
      pf.q[1] = *(const int4*)(pr + 16);
    }
#pragma unroll
    for (int t = 0; t < 4; ++t) {
      const _Float16* vp = Vb + (size_t)(16 * t + rlo) * SEQ + kt + hb8;
      FragH vf;
      vf.q[0] = *(const int4*)(vp);
      vf.q[1] = *(const int4*)(vp + 16);
      acc[t] = __builtin_amdgcn_wmma_f32_16x16x32_f16(false, pf.v, false, vf.v, (short)0, acc[t], false, false);
    }
  }
#pragma unroll
  for (int t = 0; t < 4; ++t)
#pragma unroll
    for (int j = 0; j < 8; ++j) {
      int qg = qbase + j + hb8;
      int d  = 16 * t + rlo;
      float val = acc[t][j] / lrow[j];
      O1[((size_t)(b * SEQ + qg)) * EMB + h * HD + d] = (_Float16)val;
    }
}

extern "C" void kernel_launch(void* const* d_in, const int* in_sizes, int n_in,
                              void* d_out, int out_size, void* d_ws, size_t ws_size,
                              hipStream_t stream) {
  const float* x  = (const float*)d_in[0];
  const float* wq = (const float*)d_in[1];
  const float* bq = (const float*)d_in[2];
  const float* wk = (const float*)d_in[3];
  const float* bk = (const float*)d_in[4];
  const float* wv = (const float*)d_in[5];
  const float* bv = (const float*)d_in[6];
  const float* wo = (const float*)d_in[7];
  const float* bo = (const float*)d_in[8];
  float* out = (float*)d_out;

  char* ws = (char*)d_ws;
  float*    scal     = (float*)ws;
  float*    partials = (float*)(ws + 256);
  int8_t*   qx       = (int8_t*)(ws + 8192);
  int8_t*   qw       = qx + XELEM;
  _Float16* Qh       = (_Float16*)(qw + 4 * (size_t)WELEM);
  _Float16* Kh       = Qh + XELEM;
  _Float16* Vt       = Kh + XELEM;
  _Float16* O1       = Vt + XELEM;
  int8_t*   qa       = (int8_t*)(O1 + XELEM);

  k_init_scalars<<<1, 64, 0, stream>>>(scal);
  k_absmax_f32<<<256, 256, 0, stream>>>(x, XELEM, scal);
  k_abssum_part<<<dim3(64, 4), 256, 0, stream>>>(wq, wk, wv, wo, WELEM, partials);
  k_sum_final<<<4, 64, 0, stream>>>(partials, scal);
  k_quant_x<<<4096, 256, 0, stream>>>(x, scal, qx, XELEM);
  k_quant_w<<<dim3(WELEM / 256, 4), 256, 0, stream>>>(wq, wk, wv, wo, scal, qw);
  k_gemm_qkv<<<dim3(NTOK / 128, EMB / 64, 3), 256, 0, stream>>>(
      qx, qw, bq, bk, bv, scal, Qh, Kh, Vt);
  k_flash<<<dim3(SEQ / 128, NB * NH), 256, 0, stream>>>(Qh, Kh, Vt, O1);
  k_absmax_f16<<<256, 256, 0, stream>>>(O1, XELEM, scal + 5);
  k_quant_o<<<4096, 256, 0, stream>>>(O1, scal, qa, XELEM);
  k_gemm_o<<<dim3(NTOK / 128, EMB / 64), 256, 0, stream>>>(
      qa, qw + 3 * (size_t)WELEM, bo, scal, out);
}
